// AtlasMemoryUpdate_36094905156284
// MI455X (gfx1250) — matrix operand order repaired and checked by reading, not compiled, not run on hardware
//
#include <hip/hip_runtime.h>
#include <hip/hip_bf16.h>
#include <math.h>

typedef __attribute__((ext_vector_type(16))) __bf16 v16bf;
typedef __attribute__((ext_vector_type(8)))  __bf16 v8bf;
typedef __attribute__((ext_vector_type(4)))  __bf16 v4bf;
typedef __attribute__((ext_vector_type(8)))  float  v8f;

#define NTOK 32768   // B*T
#define DDIM 768
#define HDIM 512
#define TLEN 2048
#define TILE_ELEMS (128 * 40)   // 128 rows x 32 K (+8 pad), bf16

__device__ __forceinline__ v8f wmma_bf16(v16bf a, v16bf b, v8f c) {
  return __builtin_amdgcn_wmma_f32_16x16x32_bf16(false, a, false, b, (short)0, c, false, false);
}

// ---- CDNA5 async global->LDS copy (ASYNCcnt-tracked DMA, no VGPR round-trip)
__device__ __forceinline__ void async_b128(const void* lds_dst, const void* gsrc) {
  unsigned int lds = (unsigned int)(size_t)lds_dst;           // low 32 bits = LDS offset
  unsigned long long g = (unsigned long long)(size_t)gsrc;
  asm volatile("global_load_async_to_lds_b128 %0, %1, off" ::"v"(lds), "v"(g) : "memory");
}
__device__ __forceinline__ void wait_async0() { asm volatile("s_wait_asynccnt 0" ::: "memory"); }
__device__ __forceinline__ void wait_async4() { asm volatile("s_wait_asynccnt 4" ::: "memory"); }

// Bank swizzle on 8-byte (4-elem) chunks: phys = chunk ^ ((row>>2)&6).
// Even swizzle keeps chunk-pairs contiguous -> b128 fragment loads stay legal.
__device__ __forceinline__ int swz(int row, int e) {
  return (((e >> 2) ^ ((row >> 2) & 6)) << 2);
}

__device__ __forceinline__ v16bf combine(v8bf lo, v8bf hi) {
  v16bf r;
#pragma unroll
  for (int i = 0; i < 8; ++i) { r[i] = lo[i]; r[i + 8] = hi[i]; }
  return r;
}

__device__ __forceinline__ void gelu_fwd_bwd(float x, float& g, float& dg) {
  const float kA = 0.7978845608028654f;   // sqrt(2/pi)
  const float kB = 0.044715f;
  float x2 = x * x;
  float t = tanhf(kA * (x + kB * x2 * x));
  g  = 0.5f * x * (1.0f + t);
  dg = 0.5f * (1.0f + t) + 0.5f * x * (1.0f - t * t) * kA * (1.0f + 3.0f * kB * x2);
}

// ---------------------------------------------------------------------------
// Forward-chain GEMM: C = A[M,K] * Bt^T (Bt = B^T, [N,K] row-major). Both
// tiles contiguous K-runs -> async b128 to LDS, 4-deep pipeline with
// compile-time buffer indices (all ds offsets immediate), ONE barrier/K-tile.
// Fragment LDS pointers precomputed (loop-invariant) -> no per-iter VALU, no
// WMMA WAR hazard NOPs. Tile 128x128x32, 8 waves x 8 WMMA accumulators.
// MODE 0: h = gelu(C + b1) -> out0, gelu' -> out1
// MODE 1: dpred = 2*gamma[t]*((C + b2) - values) -> out0
// MODE 2: dpre  = C * dact -> out0
// ---------------------------------------------------------------------------
template <int MODE, int K, int N>
__global__ __launch_bounds__(256) void gemm_fwd(
    const __bf16* __restrict__ A, const __bf16* __restrict__ Bt,
    const float* __restrict__ bias, const float* __restrict__ values,
    const float* __restrict__ gamma, const __bf16* __restrict__ dact,
    __bf16* __restrict__ out0, __bf16* __restrict__ out1) {
  const int tid  = threadIdx.x;
  const int lane = tid & 31;
  const int wave = tid >> 5;
  const int half = lane >> 4;
  const int ln   = lane & 15;
  const int wm   = wave & 1;   // M offset 64*wm
  const int wn   = wave >> 1;  // N offset 32*wn
  const int m0   = blockIdx.y * 128;
  const int n0   = blockIdx.x * 128;

  __shared__ __align__(16) __bf16 As[4 * TILE_ELEMS];
  __shared__ __align__(16) __bf16 Bs[4 * TILE_ELEMS];

  v8f acc[4][2];
#pragma unroll
  for (int i = 0; i < 4; ++i)
#pragma unroll
    for (int j = 0; j < 2; ++j)
#pragma unroll
      for (int e = 0; e < 8; ++e) acc[i][j][e] = 0.0f;

  // ---- loop-invariant per-thread staging descriptors (2 chunks each) ----
  int loffc[2];
  const __bf16* gAp[2];
  const __bf16* gBp[2];
#pragma unroll
  for (int i = 0; i < 2; ++i) {
    int c = tid + i * 256;                  // 512 chunks of 8 bf16
    int row = c >> 2, col = (c & 3) * 8;
    loffc[i] = row * 40 + swz(row, col);
    gAp[i] = A + (size_t)(m0 + row) * K + col;
    gBp[i] = Bt + (size_t)(n0 + row) * K + col;
  }
  // ---- loop-invariant per-lane fragment pointers (buffer 0) ----
  const __bf16* aLo[4]; const __bf16* aHi[4];
  const __bf16* bLo[2]; const __bf16* bHi[2];
#pragma unroll
  for (int im = 0; im < 4; ++im) {
    int row = wm * 64 + im * 16 + ln;
    aLo[im] = As + row * 40 + swz(row, half * 8);
    aHi[im] = As + row * 40 + swz(row, 16 + half * 8);
  }
#pragma unroll
  for (int in = 0; in < 2; ++in) {
    int row = wn * 32 + in * 16 + ln;
    bLo[in] = Bs + row * 40 + swz(row, half * 8);
    bHi[in] = Bs + row * 40 + swz(row, 16 + half * 8);
  }

  auto issue_tile = [&](int buf, int kt) {   // buf is a literal at call sites
#pragma unroll
    for (int i = 0; i < 2; ++i) {
      async_b128(As + buf * TILE_ELEMS + loffc[i], gAp[i] + kt);
      async_b128(Bs + buf * TILE_ELEMS + loffc[i], gBp[i] + kt);
    }
  };
  auto compute_tile = [&](int buf) {         // buf literal -> const ds offsets
    v16bf af[4], bfg[2];
#pragma unroll
    for (int im = 0; im < 4; ++im)
      af[im] = combine(*(const v8bf*)(aLo[im] + buf * TILE_ELEMS),
                       *(const v8bf*)(aHi[im] + buf * TILE_ELEMS));
#pragma unroll
    for (int in = 0; in < 2; ++in)
      bfg[in] = combine(*(const v8bf*)(bLo[in] + buf * TILE_ELEMS),
                        *(const v8bf*)(bHi[in] + buf * TILE_ELEMS));
#pragma unroll
    for (int im = 0; im < 4; ++im)
#pragma unroll
      for (int in = 0; in < 2; ++in)
        acc[im][in] = wmma_bf16(af[im], bfg[in], acc[im][in]);
  };

  constexpr int nk = K >> 5;                 // 24 or 16: divisible by 4
  issue_tile(0, 0);
  for (int it = 0; it < nk; it += 4) {
#pragma unroll
    for (int s = 0; s < 4; ++s) {
      int t = it + s;
      if (t + 1 < nk) {
        issue_tile((s + 1) & 3, (t + 1) << 5);   // prefetch 1 tile ahead
        wait_async4();                            // oldest 4 = current tile
      } else {
        wait_async0();
      }
      __syncthreads();                            // single barrier per K-tile
      compute_tile(s);
    }
  }

  // Epilogue: constant-folded offsets (N is compile-time).
  const int mb = m0 + wm * 64 + half * 8;
  const int nb = n0 + wn * 32 + ln;
  const int base = mb * N + nb;
#pragma unroll
  for (int im = 0; im < 4; ++im)
#pragma unroll
    for (int in = 0; in < 2; ++in)
#pragma unroll
      for (int r = 0; r < 8; ++r) {
        int idx = base + (im * 16 + r) * N + in * 16;
        float v = acc[im][in][r];
        if (MODE == 0) {
          v += bias[nb + in * 16];
          float g, dg;
          gelu_fwd_bwd(v, g, dg);
          out0[idx] = (__bf16)g;
          out1[idx] = (__bf16)dg;
        } else if (MODE == 1) {
          v += bias[nb + in * 16];
          float e = v - values[idx];                  // values ld == N == D
          float gm = gamma[(mb + im * 16 + r) & (TLEN - 1)];
          out0[idx] = (__bf16)(2.0f * gm * e);
        } else {
          v *= (float)dact[idx];                      // dact ld == N == H
          out0[idx] = (__bf16)v;
        }
      }
}

// ---------------------------------------------------------------------------
// Weight-grad GEMM: C[M,N] += sum_k A[k,m]*B[k,n]  (A:[Ktot,M], B:[Ktot,N]),
// split-K over blockIdx.z, f32 atomic accumulation (constant offsets).
// Register-prefetched 2-deep LDS pipeline, ONE barrier per tile. Transpose
// staging: v4bf global loads of 4 rows, register repack, b64 swizzled stores.
// ---------------------------------------------------------------------------
template <int M, int N, int KCH>
__global__ __launch_bounds__(256) void gemm_at(
    const __bf16* __restrict__ A, const __bf16* __restrict__ B,
    float* __restrict__ C) {
  const int tid  = threadIdx.x;
  const int lane = tid & 31;
  const int wave = tid >> 5;
  const int half = lane >> 4;
  const int ln   = lane & 15;
  const int wm   = wave & 1;
  const int wn   = wave >> 1;
  const int m0   = blockIdx.y * 128;
  const int n0   = blockIdx.x * 128;
  const int k0   = blockIdx.z * KCH;

  __shared__ __align__(16) __bf16 As[2 * TILE_ELEMS];
  __shared__ __align__(16) __bf16 Bs[2 * TILE_ELEMS];

  v8f acc[4][2];
#pragma unroll
  for (int i = 0; i < 4; ++i)
#pragma unroll
    for (int j = 0; j < 2; ++j)
#pragma unroll
      for (int e = 0; e < 8; ++e) acc[i][j][e] = 0.0f;

  // thread owns a 4k x 4x patch of each tile: 32 x-quads * 8 k-quads = 256
  const int xq = (tid & 31) * 4;
  const int kq = (tid >> 5) * 4;

  v4bf a4[4], b4[4];                         // register prefetch of one tile
  auto load_regs = [&](int kt) {
#pragma unroll
    for (int r = 0; r < 4; ++r) {
      a4[r] = *(const v4bf*)(A + (size_t)(kt + kq + r) * M + m0 + xq);
      b4[r] = *(const v4bf*)(B + (size_t)(kt + kq + r) * N + n0 + xq);
    }
  };
  auto store_tile = [&](int buf) {
#pragma unroll
    for (int e = 0; e < 4; ++e) {
      v4bf pa, pb;
#pragma unroll
      for (int r = 0; r < 4; ++r) { pa[r] = a4[r][e]; pb[r] = b4[r][e]; }
      int loff = (xq + e) * 40 + swz(xq + e, kq);
      *(v4bf*)(As + buf * TILE_ELEMS + loff) = pa;
      *(v4bf*)(Bs + buf * TILE_ELEMS + loff) = pb;
    }
  };

  // loop-invariant per-lane fragment pointers (buffer 0)
  const __bf16* aLo[4]; const __bf16* aHi[4];
  const __bf16* bLo[2]; const __bf16* bHi[2];
#pragma unroll
  for (int im = 0; im < 4; ++im) {
    int row = wm * 64 + im * 16 + ln;
    aLo[im] = As + row * 40 + swz(row, half * 8);
    aHi[im] = As + row * 40 + swz(row, 16 + half * 8);
  }
#pragma unroll
  for (int in = 0; in < 2; ++in) {
    int row = wn * 32 + in * 16 + ln;
    bLo[in] = Bs + row * 40 + swz(row, half * 8);
    bHi[in] = Bs + row * 40 + swz(row, 16 + half * 8);
  }
  auto compute_tile = [&](int buf) {
    v16bf af[4], bfg[2];
#pragma unroll
    for (int im = 0; im < 4; ++im)
      af[im] = combine(*(const v8bf*)(aLo[im] + buf * TILE_ELEMS),
                       *(const v8bf*)(aHi[im] + buf * TILE_ELEMS));
#pragma unroll
    for (int in = 0; in < 2; ++in)
      bfg[in] = combine(*(const v8bf*)(bLo[in] + buf * TILE_ELEMS),
                        *(const v8bf*)(bHi[in] + buf * TILE_ELEMS));
#pragma unroll
    for (int im = 0; im < 4; ++im)
#pragma unroll
      for (int in = 0; in < 2; ++in)
        acc[im][in] = wmma_bf16(af[im], bfg[in], acc[im][in]);
  };

  load_regs(k0);
  for (int kt = k0; kt < k0 + KCH; kt += 64) {   // KCH multiple of 64
#pragma unroll
    for (int s = 0; s < 2; ++s) {
      store_tile(s);
      __syncthreads();                       // stores visible; single barrier
      int nextk = kt + 32 * (s + 1);
      if (nextk < k0 + KCH) load_regs(nextk);  // global latency under compute
      compute_tile(s);
    }
  }

  float* Cb = C + (size_t)((m0 + wm * 64 + half * 8) * N + n0 + wn * 32 + ln);
#pragma unroll
  for (int im = 0; im < 4; ++im)
#pragma unroll
    for (int in = 0; in < 2; ++in)
#pragma unroll
      for (int r = 0; r < 8; ++r)
        atomicAdd(Cb + (im * 16 + r) * N + in * 16, acc[im][in][r]);
}

// --------------------------- small utility kernels -------------------------
__global__ void zero_f32(float* p, size_t n) {
  size_t i = (size_t)blockIdx.x * blockDim.x + threadIdx.x;
  for (; i < n; i += (size_t)gridDim.x * blockDim.x) p[i] = 0.0f;
}

__global__ void cvt_bf16(const float* __restrict__ in, __bf16* __restrict__ out, size_t n) {
  size_t i = (size_t)blockIdx.x * blockDim.x + threadIdx.x;
  for (; i < n; i += (size_t)gridDim.x * blockDim.x) out[i] = (__bf16)in[i];
}

// Tiled transpose: out[C][R] = in[R][C]^T (f32 -> bf16). Block (32,8).
__global__ void cvt_transpose(const float* __restrict__ in, __bf16* __restrict__ out,
                              int R, int Ccols) {
  __shared__ float t[32][33];
  int bc = blockIdx.x * 32;   // column block (input)
  int br = blockIdx.y * 32;   // row block (input)
  int x = threadIdx.x, y = threadIdx.y;
#pragma unroll
  for (int i = 0; i < 32; i += 8)
    t[y + i][x] = in[(size_t)(br + y + i) * Ccols + bc + x];
  __syncthreads();
#pragma unroll
  for (int i = 0; i < 32; i += 8)
    out[(size_t)(bc + y + i) * R + br + x] = (__bf16)t[x][y + i];
}

// db[n] = sum_k G[k*N + n]; v8bf loads; block = N/8 threads, grid = k-chunks
template <int N>
__global__ void colsum(const __bf16* __restrict__ G, float* __restrict__ out,
                       int kchunk) {
  int n8 = threadIdx.x * 8;
  size_t k0 = (size_t)blockIdx.x * kchunk;
  float s[8];
#pragma unroll
  for (int e = 0; e < 8; ++e) s[e] = 0.0f;
  for (size_t k = k0; k < k0 + kchunk; ++k) {
    v8bf v = *(const v8bf*)(G + k * N + n8);
#pragma unroll
    for (int e = 0; e < 8; ++e) s[e] += (float)v[e];
  }
#pragma unroll
  for (int e = 0; e < 8; ++e) atomicAdd(&out[n8 + e], s[e]);
}

__global__ void sqnorm(const float* __restrict__ g, size_t n,
                       float* __restrict__ norms, int slot) {
  __shared__ float red[256];
  float s = 0.0f;
  for (size_t i = (size_t)blockIdx.x * blockDim.x + threadIdx.x; i < n;
       i += (size_t)gridDim.x * blockDim.x) {
    float v = g[i];
    s += v * v;
  }
  red[threadIdx.x] = s;
  __syncthreads();
  for (int off = 128; off > 0; off >>= 1) {
    if (threadIdx.x < off) red[threadIdx.x] += red[threadIdx.x + off];
    __syncthreads();
  }
  if (threadIdx.x == 0) atomicAdd(&norms[slot], red[0]);
}

__global__ void finalize_m(const float* __restrict__ norms,
                           const float* __restrict__ momentum,
                           float* __restrict__ scales) {
  float m = momentum[0];
#pragma unroll
  for (int i = 0; i < 4; ++i) {
    m = 0.9f * m + 0.1f * sqrtf(norms[i]);
    scales[i] = m + 1e-8f;
  }
}

__global__ void update_params(
    const float* __restrict__ W1, const float* __restrict__ b1,
    const float* __restrict__ W2, const float* __restrict__ b2,
    const float* __restrict__ dW1, const float* __restrict__ db1,
    const float* __restrict__ dW2, const float* __restrict__ db2,
    const float* __restrict__ scales, float* __restrict__ out) {
  const int S0 = DDIM * HDIM, S1 = HDIM, S2 = HDIM * DDIM, S3 = DDIM;
  const int O1 = S0, O2 = S0 + S1, O3 = S0 + S1 + S2, TOT = S0 + S1 + S2 + S3;
  int i = blockIdx.x * blockDim.x + threadIdx.x;
  if (i >= TOT) return;
  float p, g, sc;
  if (i < O1)      { p = W1[i];      g = dW1[i];      sc = scales[0]; }
  else if (i < O2) { p = b1[i - O1]; g = db1[i - O1]; sc = scales[1]; }
  else if (i < O3) { p = W2[i - O2]; g = dW2[i - O2]; sc = scales[2]; }
  else             { p = b2[i - O3]; g = db2[i - O3]; sc = scales[3]; }
  g = fminf(0.1f, fmaxf(-0.1f, g));
  float upd = 0.01f * g / sc;
  upd = fminf(0.01f, fmaxf(-0.01f, upd));
  out[i] = p - upd;
}

// ---------------------------------------------------------------------------
extern "C" void kernel_launch(void* const* d_in, const int* in_sizes, int n_in,
                              void* d_out, int out_size, void* d_ws, size_t ws_size,
                              hipStream_t stream) {
  const float* keys     = (const float*)d_in[0];
  const float* values   = (const float*)d_in[1];
  const float* gamma    = (const float*)d_in[2];
  const float* W1       = (const float*)d_in[3];
  const float* b1       = (const float*)d_in[4];
  const float* W2       = (const float*)d_in[5];
  const float* b2       = (const float*)d_in[6];
  const float* momentum = (const float*)d_in[7];
  float* out = (float*)d_out;

  // ---- workspace carve-up (256B aligned) ----
  char* base = (char*)d_ws;
  size_t off = 0;
  auto alloc = [&](size_t bytes) -> size_t {
    size_t r = off;
    off = (off + bytes + 255) & ~(size_t)255;
    return r;
  };
  size_t o_dW1   = alloc((size_t)DDIM * HDIM * 4);
  size_t o_dW2   = alloc((size_t)HDIM * DDIM * 4);
  size_t o_db1   = alloc((size_t)HDIM * 4);
  size_t o_db2   = alloc((size_t)DDIM * 4);
  size_t o_norms = alloc(16);
  size_t o_scale = alloc(16);
  size_t o_keysb = alloc((size_t)NTOK * DDIM * 2);
  size_t o_hb    = alloc((size_t)NTOK * HDIM * 2);
  size_t o_dactb = alloc((size_t)NTOK * HDIM * 2);
  size_t o_dpred = alloc((size_t)NTOK * DDIM * 2);
  size_t o_dpre  = alloc((size_t)NTOK * HDIM * 2);
  size_t o_W1Tb  = alloc((size_t)DDIM * HDIM * 2);   // [H][D]
  size_t o_W2b   = alloc((size_t)HDIM * DDIM * 2);   // [H][D]
  size_t o_W2Tb  = alloc((size_t)DDIM * HDIM * 2);   // [D][H]
  (void)ws_size; (void)n_in; (void)in_sizes; (void)out_size;

  float*  dW1   = (float*)(base + o_dW1);
  float*  dW2   = (float*)(base + o_dW2);
  float*  db1   = (float*)(base + o_db1);
  float*  db2   = (float*)(base + o_db2);
  float*  norms = (float*)(base + o_norms);
  float*  scales= (float*)(base + o_scale);
  __bf16* keysb = (__bf16*)(base + o_keysb);
  __bf16* hb    = (__bf16*)(base + o_hb);
  __bf16* dactb = (__bf16*)(base + o_dactb);
  __bf16* dpred = (__bf16*)(base + o_dpred);
  __bf16* dpre  = (__bf16*)(base + o_dpre);
  __bf16* W1Tb  = (__bf16*)(base + o_W1Tb);
  __bf16* W2b   = (__bf16*)(base + o_W2b);
  __bf16* W2Tb  = (__bf16*)(base + o_W2Tb);

  // 1) zero grads + norms (contiguous region starting at dW1)
  size_t zn = (size_t)DDIM * HDIM + (size_t)HDIM * DDIM + HDIM + DDIM + 4;
  zero_f32<<<1024, 256, 0, stream>>>(dW1, zn);

  // 2) bf16 conversions: keys, W2 natural; W1^T, W2^T tiled transposes
  cvt_bf16<<<4096, 256, 0, stream>>>(keys, keysb, (size_t)NTOK * DDIM);
  cvt_bf16<<<512, 256, 0, stream>>>(W2, W2b, (size_t)HDIM * DDIM);
  cvt_transpose<<<dim3(HDIM / 32, DDIM / 32), dim3(32, 8), 0, stream>>>(
      W1, W1Tb, DDIM, HDIM);   // [H][D]
  cvt_transpose<<<dim3(DDIM / 32, HDIM / 32), dim3(32, 8), 0, stream>>>(
      W2, W2Tb, HDIM, DDIM);   // [D][H]

  // 3) forward: h = gelu(keys@W1 + b1); also gelu'   (Bt = W1^T)
  gemm_fwd<0, DDIM, HDIM><<<dim3(HDIM / 128, NTOK / 128), 256, 0, stream>>>(
      keysb, W1Tb, b1, nullptr, nullptr, nullptr, hb, dactb);

  // 4) dpred = 2*gamma*(h@W2 + b2 - values)          (Bt = W2^T)
  gemm_fwd<1, HDIM, DDIM><<<dim3(DDIM / 128, NTOK / 128), 256, 0, stream>>>(
      hb, W2Tb, b2, values, gamma, nullptr, dpred, nullptr);

  // 5) dpre = (dpred@W2^T) * gelu'                   (Bt = (W2^T)^T = W2)
  gemm_fwd<2, DDIM, HDIM><<<dim3(HDIM / 128, NTOK / 128), 256, 0, stream>>>(
      dpred, W2b, nullptr, nullptr, nullptr, dactb, dpre, nullptr);

  // 6) dW2 = h^T @ dpred  [H, D], split-K=16
  gemm_at<HDIM, DDIM, NTOK / 16>
      <<<dim3(DDIM / 128, HDIM / 128, 16), 256, 0, stream>>>(hb, dpred, dW2);

  // 7) dW1 = keys^T @ dpre  [D, H]
  gemm_at<DDIM, HDIM, NTOK / 16>
      <<<dim3(HDIM / 128, DDIM / 128, 16), 256, 0, stream>>>(keysb, dpre, dW1);

  // 8) bias grads (vectorized column sums)
  colsum<HDIM><<<32, HDIM / 8, 0, stream>>>(dpre, db1, NTOK / 32);
  colsum<DDIM><<<32, DDIM / 8, 0, stream>>>(dpred, db2, NTOK / 32);

  // 9) grad norms (order: W1, b1, W2, b2)
  sqnorm<<<512, 256, 0, stream>>>(dW1, (size_t)DDIM * HDIM, norms, 0);
  sqnorm<<<2, 256, 0, stream>>>(db1, (size_t)HDIM, norms, 1);
  sqnorm<<<512, 256, 0, stream>>>(dW2, (size_t)HDIM * DDIM, norms, 2);
  sqnorm<<<3, 256, 0, stream>>>(db2, (size_t)DDIM, norms, 3);

  // 10) momentum recurrence + scales
  finalize_m<<<1, 1, 0, stream>>>(norms, momentum, scales);

  // 11) clipped update -> concatenated output
  const int TOT = DDIM * HDIM + HDIM + HDIM * DDIM + DDIM;
  update_params<<<(TOT + 255) / 256, 256, 0, stream>>>(
      W1, b1, W2, b2, dW1, db1, dW2, db2, scales, out);
}
